// Cluster_71150428225981
// MI455X (gfx1250) — compile-verified, matrix operand order checked by reading, not run on hardware
//
#include <hip/hip_runtime.h>
#include <hip/hip_bf16.h>
#include <math.h>

typedef __attribute__((ext_vector_type(16))) __bf16 v16bf;
typedef __attribute__((ext_vector_type(8)))  float  v8f;

#define NSPAT 3136   // 56*56 per batch
#define W0 56

// ---------------------------------------------------------------------------
// Fused q/v 1x1-conv GEMM:  Q[batch] = Wc(96x128)*X + bc ; V = Wp*X + bp.
// One wave computes a 16x16 tile of BOTH outputs, loading the shared X (B)
// fragment once per K-step and issuing two WMMAs against it. This halves the
// HBM read traffic of x (103 MB saved) vs two separate GEMM launches.
// blockDim.x = 192 (6 waves = 96 rows / 16).
// ---------------------------------------------------------------------------
__global__ void conv_qv_wmma_bf16(const float* __restrict__ Wc,
                                  const float* __restrict__ bc,
                                  const float* __restrict__ Wp,
                                  const float* __restrict__ bp,
                                  const float* __restrict__ Xin,  // [64][128][N]
                                  float* __restrict__ Qout,       // [64][96][N]
                                  float* __restrict__ Vout)       // [64][96][N]
{
    const int N = NSPAT, K = 128;
    const int wave = threadIdx.x >> 5;
    const int lane = threadIdx.x & 31;
    const int col0 = blockIdx.x * 16;
    const int row0 = wave * 16;
    const int batch = blockIdx.y;

    const float* Xb = Xin + (size_t)batch * K * N;

    const int mr = lane & 15;          // A row / B col / D col within tile
    const int kb = (lane >> 4) << 3;   // lane-group K base: 0 or 8

    v8f accq = {}, accv = {};
    for (int k0 = 0; k0 < K; k0 += 32) {
        v16bf afc, afp, bfr;
        const float* Crow = Wc + (size_t)(row0 + mr) * K + (k0 + kb);
        const float* Prow = Wp + (size_t)(row0 + mr) * K + (k0 + kb);
        const float* Xcol = Xb + (size_t)(k0 + kb) * N + (col0 + mr);
#pragma unroll
        for (int e = 0; e < 16; ++e) {
            const int kk = ((e >> 3) << 4) + (e & 7);   // 0..7, 16..23
            afc[e] = (__bf16)Crow[kk];
            afp[e] = (__bf16)Prow[kk];
            bfr[e] = (__bf16)Xcol[(size_t)kk * N];
        }
        accq = __builtin_amdgcn_wmma_f32_16x16x32_bf16(false, afc, false, bfr,
                                                       (short)0, accq, false, false);
        accv = __builtin_amdgcn_wmma_f32_16x16x32_bf16(false, afp, false, bfr,
                                                       (short)0, accv, false, false);
    }

    // D layout: VGPR r -> M = r + 8*(lane>=16), N = lane&15
    const int half = lane >> 4;
    const int n = col0 + mr;
    float* Qb = Qout + (size_t)batch * 96 * N;
    float* Vb = Vout + (size_t)batch * 96 * N;
#pragma unroll
    for (int r = 0; r < 8; ++r) {
        const int row = row0 + r + half * 8;
        Qb[(size_t)row * N + n] = accq[r] + bc[row];
        Vb[(size_t)row * N + n] = accv[r] + bp[row];
    }
}

// ---------------------------------------------------------------------------
// Generic per-batch 1x1-conv GEMM:  Out[batch] = A(OxK) * X[batch](KxN) + bias
// Used for the output projection (O=128, K=96). blockDim.x = 32*(O/16).
// ---------------------------------------------------------------------------
__global__ void conv1x1_wmma_bf16(const float* __restrict__ A,
                                  const float* __restrict__ bias,
                                  const float* __restrict__ Xin,  // [nb][K][N]
                                  float* __restrict__ Out,        // [nb][O][N]
                                  int O, int K)
{
    const int N = NSPAT;
    const int wave = threadIdx.x >> 5;
    const int lane = threadIdx.x & 31;
    const int col0 = blockIdx.x * 16;
    const int row0 = wave * 16;
    const int batch = blockIdx.y;

    const float* Xb = Xin + (size_t)batch * K * N;
    float*       Ob = Out + (size_t)batch * O * N;

    const int mr = lane & 15;
    const int kb = (lane >> 4) << 3;

    v8f acc = {};
    for (int k0 = 0; k0 < K; k0 += 32) {
        v16bf afr, bfr;
        const float* Arow = A  + (size_t)(row0 + mr) * K + (k0 + kb);
        const float* Xcol = Xb + (size_t)(k0 + kb) * N + (col0 + mr);
#pragma unroll
        for (int e = 0; e < 16; ++e) {
            const int kk = ((e >> 3) << 4) + (e & 7);
            afr[e] = (__bf16)Arow[kk];
            bfr[e] = (__bf16)Xcol[(size_t)kk * N];
        }
        acc = __builtin_amdgcn_wmma_f32_16x16x32_bf16(false, afr, false, bfr,
                                                      (short)0, acc, false, false);
    }

    const int half = lane >> 4;
    const int n = col0 + mr;
#pragma unroll
    for (int r = 0; r < 8; ++r) {
        const int row = row0 + r + half * 8;
        Ob[(size_t)row * N + n] = acc[r] + bias[row];
    }
}

// ---------------------------------------------------------------------------
// Clustering stage: one workgroup per fold block b in [0,1024).
//   b = ((batch*4 + head)*2 + f1)*2 + f2 ; tile = 24 ch x 28 x 28
// Reads q,v (layout [64][96][56][56]); writes patches IN PLACE over q.
// ---------------------------------------------------------------------------
__global__ void cluster_stage(float* __restrict__ qb,        // q in, patches out
                              const float* __restrict__ vb,  // value
                              const float* __restrict__ alpha_p,
                              const float* __restrict__ beta_p)
{
    __shared__ float s_cent[4][24];    // centers (normalized in place)
    __shared__ float s_vcent[4][24];   // value centers (raw means)
    __shared__ float s_clus[4][24];    // csum -> cluster
    __shared__ float s_ssum[4];
    __shared__ float s_s[784];
    __shared__ int   s_idx[784];
    __shared__ float s_alpha, s_beta;

    const int tid   = threadIdx.x;
    const int blk   = blockIdx.x;
    const int f2    = blk & 1;
    const int f1    = (blk >> 1) & 1;
    const int head  = (blk >> 2) & 3;
    const int batch = blk >> 4;

    const size_t base0 = ((size_t)batch * 96 + head * 24) * NSPAT
                       + (size_t)(f1 * 28) * W0 + (f2 * 28);

    if (tid == 0) { s_alpha = alpha_p[0]; s_beta = beta_p[0]; }
    if (tid < 4)  s_ssum[tid] = 0.f;

    // ---- centers: mean over 14x14 blocks (96 (m,c) pairs) ----
    if (tid < 96) {
        const int m = tid / 24, c = tid % 24;
        const int pw = m >> 1, ph = m & 1;
        const size_t cb = base0 + (size_t)c * NSPAT;
        float sq = 0.f, sv = 0.f;
        for (int i = 0; i < 196; ++i) {
            const int w = pw * 14 + i / 14;
            const int h = ph * 14 + i % 14;
            const size_t off = cb + (size_t)w * W0 + h;
            sq += qb[off];
            sv += vb[off];
        }
        s_cent[m][c]  = sq * (1.f / 196.f);
        s_vcent[m][c] = sv * (1.f / 196.f);
        s_clus[m][c]  = 0.f;
    }
    __syncthreads();

    // ---- normalize centers ----
    if (tid < 4) {
        float nn = 0.f;
        for (int c = 0; c < 24; ++c) nn += s_cent[tid][c] * s_cent[tid][c];
        const float inv = 1.f / fmaxf(sqrtf(nn), 1e-12f);
        for (int c = 0; c < 24; ++c) s_cent[tid][c] *= inv;
    }
    __syncthreads();

    // ---- per-position assignment + weighted accumulation ----
    for (int n = tid; n < 784; n += blockDim.x) {
        const int w = n / 28, h = n % 28;
        const size_t off0 = base0 + (size_t)w * W0 + h;
        float qv[24], vv[24];
        float qn = 0.f;
#pragma unroll
        for (int c = 0; c < 24; ++c) {
            const float t = qb[off0 + (size_t)c * NSPAT];
            qv[c] = t; qn += t * t;
            vv[c] = vb[off0 + (size_t)c * NSPAT];
        }
        const float invq = 1.f / fmaxf(sqrtf(qn), 1e-12f);
        float best = -1e30f; int bi = 0;
#pragma unroll
        for (int m = 0; m < 4; ++m) {
            float d = 0.f;
#pragma unroll
            for (int c = 0; c < 24; ++c) d += s_cent[m][c] * qv[c];
            const float sim = 1.f / (1.f + expf(-(s_beta + s_alpha * d * invq)));
            if (sim > best) { best = sim; bi = m; }   // first-index tie-break
        }
        s_s[n] = best; s_idx[n] = bi;
        atomicAdd(&s_ssum[bi], best);
#pragma unroll
        for (int c = 0; c < 24; ++c) atomicAdd(&s_clus[bi][c], best * vv[c]);
    }
    __syncthreads();

    // ---- cluster means ----
    if (tid < 96) {
        const int m = tid / 24, c = tid % 24;
        s_clus[m][c] = (s_clus[m][c] + s_vcent[m][c]) / (s_ssum[m] + 1.f);
    }
    __syncthreads();

    // ---- scatter patches (overwrites q buffer; all q reads are done) ----
    for (int n = tid; n < 784; n += blockDim.x) {
        const int w = n / 28, h = n % 28;
        const size_t off0 = base0 + (size_t)w * W0 + h;
        const float s = s_s[n];
        const int bi = s_idx[n];
#pragma unroll
        for (int c = 0; c < 24; ++c)
            qb[off0 + (size_t)c * NSPAT] = s * s_clus[bi][c];
    }
}

// ---------------------------------------------------------------------------
extern "C" void kernel_launch(void* const* d_in, const int* in_sizes, int n_in,
                              void* d_out, int out_size, void* d_ws, size_t ws_size,
                              hipStream_t stream) {
    const float* x     = (const float*)d_in[0];   // [64,128,56,56]
    const float* Wc    = (const float*)d_in[1];   // [96,128]
    const float* bc    = (const float*)d_in[2];   // [96]
    const float* Wp    = (const float*)d_in[3];   // [96,128]
    const float* bp    = (const float*)d_in[4];   // [96]
    const float* alpha = (const float*)d_in[5];   // [1]
    const float* beta  = (const float*)d_in[6];   // [1]
    const float* Wo    = (const float*)d_in[7];   // [128,96]
    const float* bo    = (const float*)d_in[8];   // [128]
    float* out = (float*)d_out;                   // [64,128,56,56]

    float* qbuf = (float*)d_ws;                         // 64*96*3136 f32
    float* vbuf = qbuf + (size_t)64 * 96 * NSPAT;       // 64*96*3136 f32

    // fused q/v conv: x read once, two WMMAs per B fragment
    dim3 gq(NSPAT / 16, 64);
    conv_qv_wmma_bf16<<<gq, 192, 0, stream>>>(Wc, bc, Wp, bp, x, qbuf, vbuf);

    // clustering + scatter (patches written over qbuf)
    cluster_stage<<<1024, 256, 0, stream>>>(qbuf, vbuf, alpha, beta);

    // out = Wo * patches + bo   (O=128, K=96 -> 8 waves/block)
    conv1x1_wmma_bf16<<<gq, 256, 0, stream>>>(Wo, bo, qbuf, out, 128, 96);
}
